// GTN_16587163697420
// MI455X (gfx1250) — compile-verified, matrix operand order checked by reading, not run on hardware
//
#include <hip/hip_runtime.h>
#include <hip/hip_bf16.h>

typedef __attribute__((ext_vector_type(16))) __bf16 v16bf;
typedef __attribute__((ext_vector_type(8)))  __bf16 v8bf;
typedef __attribute__((ext_vector_type(8)))  float  v8f;

#define N_NODES 100000
#define N_EDGES 400000
#define HIDDIM  128
#define HEADS   4
#define CH      32
#define ATT_SCALE 0.17677669529663687f   // 1/sqrt(32)

#define BM      64      // rows per block in the projection GEMM
#define LDSTR   136     // LDS row stride in bf16 (272B: 16B-aligned, bank-spread)

// ---- monotonic float<->uint encoding for atomic float max ------------------
__device__ __forceinline__ unsigned enc_f(float f) {
    unsigned u = __float_as_uint(f);
    return (u >> 31) ? ~u : (u | 0x80000000u);
}
__device__ __forceinline__ float dec_f(unsigned e) {
    return __uint_as_float((e >> 31) ? (e & 0x7FFFFFFFu) : ~e);
}

// ---- init: zero aggregate + denom, set running max to -inf -----------------
__global__ void gtn_init(float* __restrict__ aggr, float* __restrict__ denom,
                         unsigned* __restrict__ maxu) {
    int i = blockIdx.x * blockDim.x + threadIdx.x;
    if (i < N_NODES * HIDDIM) aggr[i] = 0.0f;
    if (i < N_NODES * HEADS) { denom[i] = 0.0f; maxu[i] = enc_f(-1e30f); }
}

// ---- fused 4-matrix projection GEMM via v_wmma_f32_16x16x32_bf16 -----------
// H: N x 128 f32 row-major.  W: 128 x 128 row-major [k*128+n].
// Block tile: 64 rows x 128 cols. 8 waves; wave w owns 16 cols, 4 M-subtiles.
// A staged 64x128 bf16 in LDS; W staged transposed (sW[n][k]) bf16 in LDS.
__global__ __launch_bounds__(256)
void gtn_gemm4(const float* __restrict__ H,
               const float* __restrict__ W0, const float* __restrict__ W1,
               const float* __restrict__ W2, const float* __restrict__ W3,
               const float* __restrict__ b0, const float* __restrict__ b1,
               const float* __restrict__ b2, const float* __restrict__ b3,
               float* __restrict__ O0, float* __restrict__ O1,
               float* __restrict__ O2, float* __restrict__ O3) {
    const float* W; const float* B; float* O;
    switch (blockIdx.y) {
        case 0:  W = W0; B = b0; O = O0; break;
        case 1:  W = W1; B = b1; O = O1; break;
        case 2:  W = W2; B = b2; O = O2; break;
        default: W = W3; B = b3; O = O3; break;
    }
    __shared__ __bf16 sA[BM * LDSTR];        // 17.4 KB
    __shared__ __bf16 sW[HIDDIM * LDSTR];    // 34.8 KB
    const int row0 = blockIdx.x * BM;

    // stage A tile (64 x 128), f32 -> bf16, float4-granular, row-guarded
    for (int i = threadIdx.x; i < BM * 32; i += 256) {
        const int r = i >> 5, c4 = i & 31;
        const int row = row0 + r;
        float4 v = make_float4(0.f, 0.f, 0.f, 0.f);
        if (row < N_NODES) v = ((const float4*)(H + (size_t)row * HIDDIM))[c4];
        __bf16* dst = sA + r * LDSTR + c4 * 4;
        dst[0] = (__bf16)v.x; dst[1] = (__bf16)v.y;
        dst[2] = (__bf16)v.z; dst[3] = (__bf16)v.w;
    }
    // stage W transposed: sW[n*LDSTR + k] = W[k*128 + n]  (coalesced reads)
    for (int i = threadIdx.x; i < HIDDIM * 32; i += 256) {
        const int k = i >> 5, n4 = i & 31;
        const float4 v = ((const float4*)(W + (size_t)k * HIDDIM))[n4];
        const int n = n4 * 4;
        sW[(n + 0) * LDSTR + k] = (__bf16)v.x;
        sW[(n + 1) * LDSTR + k] = (__bf16)v.y;
        sW[(n + 2) * LDSTR + k] = (__bf16)v.z;
        sW[(n + 3) * LDSTR + k] = (__bf16)v.w;
    }
    __syncthreads();

    const int lane = threadIdx.x & 31;
    const int wave = threadIdx.x >> 5;
    const int m    = lane & 15;               // A row within 16-row subtile
    const int grp  = (lane >> 4) & 1;         // lane group -> K +8 offset
    const int ncol = wave * 16 + (lane & 15); // output column (0..127)

    v8f acc[4];
    {
        const float bias = B[ncol];
        #pragma unroll
        for (int mt = 0; mt < 4; ++mt)
            #pragma unroll
            for (int j = 0; j < 8; ++j) acc[mt][j] = bias;
    }

    #pragma unroll
    for (int ks = 0; ks < 4; ++ks) {          // K = 128 in steps of 32
        const int kb = ks * 32 + grp * 8;
        // B fragment: two contiguous 16B runs from transposed W tile
        const v8bf blo = *(const v8bf*)(sW + (size_t)ncol * LDSTR + kb);
        const v8bf bhi = *(const v8bf*)(sW + (size_t)ncol * LDSTR + kb + 16);
        v16bf b;
        #pragma unroll
        for (int e = 0; e < 8; ++e) { b[e] = blo[e]; b[e + 8] = bhi[e]; }
        #pragma unroll
        for (int mt = 0; mt < 4; ++mt) {
            const __bf16* arow = sA + (mt * 16 + m) * LDSTR;
            const v8bf alo = *(const v8bf*)(arow + kb);
            const v8bf ahi = *(const v8bf*)(arow + kb + 16);
            v16bf a;
            #pragma unroll
            for (int e = 0; e < 8; ++e) { a[e] = alo[e]; a[e + 8] = ahi[e]; }
            acc[mt] = __builtin_amdgcn_wmma_f32_16x16x32_bf16(
                false, a, false, b, (short)0, acc[mt], false, false);
        }
    }
    // C/D layout: lanes 0-15 -> rows 0..7 (vgpr j), lanes 16-31 -> rows 8..15
    #pragma unroll
    for (int mt = 0; mt < 4; ++mt) {
        #pragma unroll
        for (int j = 0; j < 8; ++j) {
            const int row = row0 + mt * 16 + j + grp * 8;
            if (row < N_NODES) O[(size_t)row * HIDDIM + ncol] = acc[mt][j];
        }
    }
}

// ---- attention logits + segment max (per edge, per head) -------------------
__global__ void gtn_alpha(const float* __restrict__ qn, const float* __restrict__ kn,
                          const float* __restrict__ ea, const float* __restrict__ We,
                          const int* __restrict__ ei,
                          float* __restrict__ ex, unsigned* __restrict__ maxu) {
    int id = blockIdx.x * blockDim.x + threadIdx.x;
    if (id >= N_EDGES * HEADS) return;
    const int e = id >> 2, h = id & 3;
    const int s = ei[e];
    const int d = ei[N_EDGES + e];
    const float a = ea[e];
    const float4* q4 = (const float4*)(qn + (size_t)d * HIDDIM + h * CH);
    const float4* k4 = (const float4*)(kn + (size_t)s * HIDDIM + h * CH);
    const float4* w4 = (const float4*)(We + h * CH);
    __builtin_prefetch(q4, 0, 0);
    __builtin_prefetch(k4, 0, 0);
    float dot = 0.0f;
    #pragma unroll
    for (int c = 0; c < 8; ++c) {
        float4 q = q4[c], k = k4[c], w = w4[c];
        dot += q.x * (k.x + a * w.x) + q.y * (k.y + a * w.y)
             + q.z * (k.z + a * w.z) + q.w * (k.w + a * w.w);
    }
    const float al = dot * ATT_SCALE;
    ex[id] = al;
    atomicMax(maxu + (size_t)d * HEADS + h, enc_f(al));
}

// ---- exp(alpha - max) + segment sum ----------------------------------------
__global__ void gtn_exp(const int* __restrict__ ei, float* __restrict__ ex,
                        const unsigned* __restrict__ maxu, float* __restrict__ denom) {
    int id = blockIdx.x * blockDim.x + threadIdx.x;
    if (id >= N_EDGES * HEADS) return;
    const int e = id >> 2, h = id & 3;
    const int d = ei[N_EDGES + e];
    const float m = dec_f(maxu[(size_t)d * HEADS + h]);
    const float v = __expf(ex[id] - m);
    ex[id] = v;
    atomicAdd(denom + (size_t)d * HEADS + h, v);
}

// ---- normalized message scatter-add ----------------------------------------
__global__ void gtn_scatter(const float* __restrict__ vn, const float* __restrict__ ex,
                            const float* __restrict__ denom, const float* __restrict__ ea,
                            const float* __restrict__ We, const int* __restrict__ ei,
                            float* __restrict__ aggr) {
    int id = blockIdx.x * blockDim.x + threadIdx.x;
    if (id >= N_EDGES * HEADS * 8) return;
    const int q = id & 7;             // channel quad within head
    const int h = (id >> 3) & 3;
    const int e = id >> 5;
    const int s = ei[e];
    const int d = ei[N_EDGES + e];
    const float a = ea[e];
    const float w = ex[(size_t)e * HEADS + h] /
                    (denom[(size_t)d * HEADS + h] + 1e-16f);
    const int c0 = h * CH + q * 4;
    const float4 v  = *(const float4*)(vn + (size_t)s * HIDDIM + c0);
    const float4 we = *(const float4*)(We + c0);
    float* o = aggr + (size_t)d * HIDDIM + c0;
    atomicAdd(o + 0, (v.x + a * we.x) * w);
    atomicAdd(o + 1, (v.y + a * we.y) * w);
    atomicAdd(o + 2, (v.z + a * we.z) * w);
    atomicAdd(o + 3, (v.w + a * we.w) * w);
}

// ---- beta-gated skip + optional LayerNorm+ReLU (one block per node) --------
__global__ __launch_bounds__(128)
void gtn_beta(const float* __restrict__ aggr, const float* __restrict__ sn,
              const float* __restrict__ Wb, const float* __restrict__ lng,
              const float* __restrict__ lnb, int do_ln, float* __restrict__ out) {
    const int n = blockIdx.x, t = threadIdx.x;
    const float o  = aggr[(size_t)n * HIDDIM + t];
    const float xr = sn[(size_t)n * HIDDIM + t];
    __shared__ float red[128];
    red[t] = o * Wb[t] + xr * Wb[128 + t] + (o - xr) * Wb[256 + t];
    __syncthreads();
    for (int s = 64; s > 0; s >>= 1) { if (t < s) red[t] += red[t + s]; __syncthreads(); }
    const float beta = 1.0f / (1.0f + __expf(-red[0]));
    __syncthreads();
    float hv = beta * xr + (1.0f - beta) * o;
    if (do_ln) {
        red[t] = hv; __syncthreads();
        for (int s = 64; s > 0; s >>= 1) { if (t < s) red[t] += red[t + s]; __syncthreads(); }
        const float mu = red[0] * (1.0f / 128.0f);
        __syncthreads();
        const float dv = hv - mu;
        red[t] = dv * dv; __syncthreads();
        for (int s = 64; s > 0; s >>= 1) { if (t < s) red[t] += red[t + s]; __syncthreads(); }
        const float var = red[0] * (1.0f / 128.0f);
        hv = dv * __frsqrt_rn(var + 1e-5f) * lng[t] + lnb[t];
        hv = fmaxf(hv, 0.0f);
    }
    out[(size_t)n * HIDDIM + t] = hv;
}

extern "C" void kernel_launch(void* const* d_in, const int* in_sizes, int n_in,
                              void* d_out, int out_size, void* d_ws, size_t ws_size,
                              hipStream_t stream) {
    const float* x  = (const float*)d_in[0];
    const int*   ei = (const int*)d_in[1];      // edge_index, (2, E)
    const float* ea = (const float*)d_in[2];    // edge_attr, (E, 1)

    const size_t NH = (size_t)N_NODES * HIDDIM;
    float* ws = (float*)d_ws;
    size_t off = 0;
    float* qn    = ws + off; off += NH;
    float* kn    = ws + off; off += NH;
    float* vn    = ws + off; off += NH;
    float* sn    = ws + off; off += NH;
    float* aggr  = ws + off; off += NH;
    float* hA    = ws + off; off += NH;
    float* hB    = ws + off; off += NH;
    unsigned* maxu = (unsigned*)(ws + off); off += (size_t)N_NODES * HEADS;
    float* denom = ws + off; off += (size_t)N_NODES * HEADS;
    float* exbuf = ws + off; off += (size_t)N_EDGES * HEADS;
    (void)ws_size; (void)in_sizes; (void)n_in; (void)out_size;

    const float* hin = x;
    float* houts[3] = { hA, hB, (float*)d_out };

    const int gemm_gx = (N_NODES + BM - 1) / BM;

    for (int l = 0; l < 3; ++l) {
        const int pb = 3 + l * 10;                   // conv{l} param base
        const float* Wq = (const float*)d_in[pb + 0];
        const float* bq = (const float*)d_in[pb + 1];
        const float* Wk = (const float*)d_in[pb + 2];
        const float* bk = (const float*)d_in[pb + 3];
        const float* Wv = (const float*)d_in[pb + 4];
        const float* bv = (const float*)d_in[pb + 5];
        const float* We = (const float*)d_in[pb + 6];
        const float* Ws = (const float*)d_in[pb + 7];
        const float* bs = (const float*)d_in[pb + 8];
        const float* Wb = (const float*)d_in[pb + 9];
        const float* lg = (l < 2) ? (const float*)d_in[33 + l * 2] : nullptr;
        const float* lb = (l < 2) ? (const float*)d_in[34 + l * 2] : nullptr;

        gtn_init<<<(N_NODES * HIDDIM + 255) / 256, 256, 0, stream>>>(aggr, denom, maxu);
        gtn_gemm4<<<dim3(gemm_gx, 4), 256, 0, stream>>>(
            hin, Wq, Wk, Wv, Ws, bq, bk, bv, bs, qn, kn, vn, sn);
        gtn_alpha<<<(N_EDGES * HEADS + 255) / 256, 256, 0, stream>>>(
            qn, kn, ea, We, ei, exbuf, maxu);
        gtn_exp<<<(N_EDGES * HEADS + 255) / 256, 256, 0, stream>>>(
            ei, exbuf, maxu, denom);
        gtn_scatter<<<(N_EDGES * HEADS * 8 + 255) / 256, 256, 0, stream>>>(
            vn, exbuf, denom, ea, We, ei, aggr);
        gtn_beta<<<N_NODES, 128, 0, stream>>>(
            aggr, sn, Wb, lg, lb, (l < 2) ? 1 : 0, houts[l]);

        hin = houts[l];
    }
}
